// PointNetPlusPlus_88527865905303
// MI455X (gfx1250) — matrix operand order repaired and checked by reading, not compiled, not run on hardware
//
#include <hip/hip_runtime.h>
#include <hip/hip_bf16.h>
#include <stdint.h>
#include <stddef.h>

// ---------------------------------------------------------------------------
// PointNet++ (SSG) forward for MI455X / gfx1250, wave32, WMMA bf16.
// All GEMM layers run on v_wmma_f32_16x16x32_bf16 (bf16 in, f32 accum).
// Weights are pre-packed per-call into per-lane B-fragment order so each
// k-step is two contiguous 16B loads per lane for A and for B.
// Each wave owns a 16(M) x 64(N) strip: one A-fragment load feeds 4 WMMAs.
// Npad is always a multiple of 64 (zero-padded in the weight pack), so the
// 4-tile strip is always full -> accumulators stay in fixed VGPRs.
// ---------------------------------------------------------------------------

typedef __bf16 v16bf __attribute__((ext_vector_type(16)));
typedef float  v8f   __attribute__((ext_vector_type(8)));
typedef unsigned int u32x4 __attribute__((ext_vector_type(4)));
typedef unsigned int u32x2 __attribute__((ext_vector_type(2)));

struct Frag32B { u32x4 lo, hi; };

static __device__ __forceinline__ v16bf ld_frag(const u32x4* plo, const u32x4* phi) {
  Frag32B t;
  t.lo = *plo;
  t.hi = *phi;
  return __builtin_bit_cast(v16bf, t);
}

// ---------------------------------------------------------------------------
// Farthest point sampling: one block per batch element, serial over samples,
// block-parallel distance update + argmax reduction (ties -> smallest index,
// matching jnp.argmax). Writes the sampled points directly.
// ---------------------------------------------------------------------------
__global__ __launch_bounds__(256)
void fps_kernel(const float* __restrict__ pts_all, int N, int S,
                float* __restrict__ out_all) {
  const int b = blockIdx.x;
  const float* pts = pts_all + (size_t)b * N * 3;
  float* out = out_all + (size_t)b * S * 3;
  __shared__ float dists[4096];
  __shared__ float rv[256];
  __shared__ int   ri[256];
  __shared__ int   s_last;
  const int tid = threadIdx.x;
  for (int i = tid; i < N; i += 256) dists[i] = 1e10f;
  if (tid == 0) { out[0] = pts[0]; out[1] = pts[1]; out[2] = pts[2]; }
  int last = 0;
  __syncthreads();
  for (int s = 1; s < S; ++s) {
    const float lx = pts[last*3+0], ly = pts[last*3+1], lz = pts[last*3+2];
    float bestv = -1.0f; int besti = 0;
    for (int i = tid; i < N; i += 256) {
      float dx = pts[i*3+0]-lx, dy = pts[i*3+1]-ly, dz = pts[i*3+2]-lz;
      float d = dx*dx + dy*dy + dz*dz;
      float dd = dists[i];
      dd = d < dd ? d : dd;
      dists[i] = dd;
      if (dd > bestv) { bestv = dd; besti = i; }   // i ascending -> first max kept
    }
    rv[tid] = bestv; ri[tid] = besti;
    __syncthreads();
    for (int off = 128; off > 0; off >>= 1) {
      if (tid < off) {
        float v2 = rv[tid+off]; int i2 = ri[tid+off];
        if (v2 > rv[tid] || (v2 == rv[tid] && i2 < ri[tid])) { rv[tid] = v2; ri[tid] = i2; }
      }
      __syncthreads();
    }
    if (tid == 0) {
      s_last = ri[0];
      out[s*3+0] = pts[ri[0]*3+0];
      out[s*3+1] = pts[ri[0]*3+1];
      out[s*3+2] = pts[ri[0]*3+2];
    }
    __syncthreads();
    last = s_last;
  }
}

// ---------------------------------------------------------------------------
// Ball query: one wave32 per center; scan points in index order, append the
// first K in-radius indices (ballot + prefix-popc), pad with first valid.
// ---------------------------------------------------------------------------
__global__ __launch_bounds__(256)
void ballq_kernel(const float* __restrict__ pts_all, int N,
                  const float* __restrict__ ctr_all, int S, int K, float r2,
                  int* __restrict__ out_all) {
  const int b = blockIdx.y;
  const int w = threadIdx.x >> 5;
  const int c = blockIdx.x * 8 + w;
  if (c >= S) return;
  const int lane = threadIdx.x & 31;
  const float* pts = pts_all + (size_t)b * N * 3;
  const float* ctr = ctr_all + ((size_t)b * S + c) * 3;
  int* out = out_all + ((size_t)b * S + c) * (size_t)K;
  const float cx = ctr[0], cy = ctr[1], cz = ctr[2];
  int filled = 0;
  int first = 0;
  bool haveFirst = false;
  for (int base = 0; base < N && filled < K; base += 32) {
    const int i = base + lane;
    bool in = false;
    if (i < N) {
      float dx = pts[i*3+0]-cx, dy = pts[i*3+1]-cy, dz = pts[i*3+2]-cz;
      in = (dx*dx + dy*dy + dz*dz) < r2;
    }
    unsigned mask = (unsigned)__ballot(in);          // wave32: low 32 bits
    int cnt = __popc(mask);
    unsigned below = (lane == 0) ? 0u : (mask & ((1u << lane) - 1u));
    int rank = __popc(below);
    if (in) {
      int slot = filled + rank;
      if (slot < K) out[slot] = i;
    }
    if (!haveFirst && cnt > 0) { first = base + __builtin_ctz(mask); haveFirst = true; }
    filled += cnt;
    if (filled > K) filled = K;
  }
  for (int s2 = filled + lane; s2 < K; s2 += 32) out[s2] = first;
}

// ---------------------------------------------------------------------------
// Level-0 grouping: relative coords only (Cin=3).
// ---------------------------------------------------------------------------
__global__ __launch_bounds__(256)
void group0_kernel(const float* __restrict__ p0, const float* __restrict__ p1,
                   const int* __restrict__ nb0, int Mrows,
                   float* __restrict__ Xc0) {
  const int row = blockIdx.x * 256 + threadIdx.x;
  if (row >= Mrows) return;
  const int sc = row >> 5;          // b*512+s
  const int b  = sc >> 9;
  const int idx = nb0[row];
  const float* src = p0 + ((size_t)b * 1024 + idx) * 3;
  const float* cen = p1 + (size_t)sc * 3;
  Xc0[(size_t)row*3+0] = src[0] - cen[0];
  Xc0[(size_t)row*3+1] = src[1] - cen[1];
  Xc0[(size_t)row*3+2] = src[2] - cen[2];
}

// ---------------------------------------------------------------------------
// Level-1 grouping: gather 128 bf16 features + relative coords.
// One "slice" thread per (row, 4 channels); t==0 also writes coords.
// ---------------------------------------------------------------------------
__global__ __launch_bounds__(256)
void group1_kernel(const float* __restrict__ p1, const float* __restrict__ p2,
                   const __bf16* __restrict__ f0, const int* __restrict__ nb1,
                   int Mrows, __bf16* __restrict__ X1, float* __restrict__ Xc1) {
  const int gid = blockIdx.x * 256 + threadIdx.x;
  const int t = gid & 31;
  const int row = gid >> 5;
  if (row >= Mrows) return;
  const int sc = row >> 6;          // b*128+s
  const int b  = sc >> 7;
  const int idx = nb1[row];
  const u32x2* src = (const u32x2*)(f0 + ((size_t)b * 512 + idx) * 128);
  u32x2* dst = (u32x2*)(X1 + (size_t)row * 128);
  dst[t] = src[t];
  if (t == 0) {
    const float* sp = p1 + ((size_t)b * 512 + idx) * 3;
    const float* cp = p2 + (size_t)sc * 3;
    Xc1[(size_t)row*3+0] = sp[0] - cp[0];
    Xc1[(size_t)row*3+1] = sp[1] - cp[1];
    Xc1[(size_t)row*3+2] = sp[2] - cp[2];
  }
}

// ---------------------------------------------------------------------------
// Weight pack: f32 W[K,N] -> bf16 fragments, layout [nt][ks][lane][16] so a
// lane reads 32 contiguous bytes per k-step. Zero-pads N up to Npad.
// B-fragment convention: lanes 0-15 hold K=ks*32+{0..7,16..23} (col = lane&15),
// lanes 16-31 hold K=ks*32+{8..15,24..31}.
// ---------------------------------------------------------------------------
__global__ __launch_bounds__(256)
void pack_w_kernel(const float* __restrict__ W, int K, int N, int Npad,
                   __bf16* __restrict__ Wp) {
  const int Ksteps = K >> 5;
  const int total = (Npad >> 4) * Ksteps * 512;
  for (int gid = blockIdx.x * 256 + threadIdx.x; gid < total; gid += gridDim.x * 256) {
    const int j    = gid & 15;
    const int lane = (gid >> 4) & 31;
    const int blk  = gid >> 9;
    const int ks   = blk % Ksteps;
    const int nt   = blk / Ksteps;
    const int nn = nt * 16 + (lane & 15);
    const int kk = ks * 32 + ((j >> 3) << 4) + ((lane >> 4) << 3) + (j & 7);
    float v = (nn < N) ? W[(size_t)kk * N + nn] : 0.0f;
    Wp[gid] = (__bf16)v;
  }
}

// ---------------------------------------------------------------------------
// First layer of mlp0 (Cin=3): plain VALU, writes bf16 activations.
// ---------------------------------------------------------------------------
__global__ __launch_bounds__(256)
void coord_mlp_kernel(const float* __restrict__ Xc, const float* __restrict__ W,
                      const float* __restrict__ b, const float* __restrict__ g,
                      const float* __restrict__ be, int M, __bf16* __restrict__ Y) {
  const int gid = blockIdx.x * 256 + threadIdx.x;
  if (gid >= M * 64) return;
  const int m = gid >> 6, n = gid & 63;
  const float* c = Xc + (size_t)m * 3;
  float v = c[0]*W[n] + c[1]*W[64+n] + c[2]*W[128+n];
  v = (v + b[n]) * g[n] + be[n];
  Y[(size_t)m * 64 + n] = (__bf16)(v > 0.f ? v : 0.f);
}

// ---------------------------------------------------------------------------
// WMMA GEMM + bias + folded-BN affine + ReLU.
//   X : bf16 [M,K]   (K multiple of 32, rows 16B aligned)
//   Wp: packed bf16 fragments (see pack_w_kernel), Npad multiple of 64
//   Xc/Wc: optional 3-channel coordinate contribution added pre-affine
//   Output bf16 (Yb) or f32 (Yf), stride Ystride, cols < Nout only.
// One wave per 16x64 strip: per k-step one A-fragment load + 4 B loads +
// 4x v_wmma_f32_16x16x32_bf16 (A reused across the 4 N-tiles).
// All accumulator indices are compile-time constants (no v_movrel).
// ---------------------------------------------------------------------------
__global__ __launch_bounds__(256)
void wmma_gemm_kernel(const __bf16* __restrict__ X, const __bf16* __restrict__ Wp,
                      const float* __restrict__ bias, const float* __restrict__ gain,
                      const float* __restrict__ beta,
                      const float* __restrict__ Xc, const float* __restrict__ Wc,
                      int M, int K, int Nout, int Npad, int Ystride,
                      __bf16* __restrict__ Yb, float* __restrict__ Yf) {
  const int Mtiles  = M >> 4;
  const int Ngroups = Npad >> 6;        // Npad is a multiple of 64
  const int Ksteps  = K >> 5;
  const int wid = (int)((blockIdx.x * 256u + (unsigned)threadIdx.x) >> 5);
  if (wid >= Mtiles * Ngroups) return;  // uniform per wave -> EXEC stays all-ones
  const int mt  = wid % Mtiles;
  const int nt0 = (wid / Mtiles) * 4;
  const int lane = threadIdx.x & 31;
  const int r  = lane & 15;
  const int hi = lane >> 4;

  // A fragment (16x32 bf16): lane<16 row=r holds K {kb..kb+7, kb+16..kb+23},
  // kb = hi*8 -> two 16B loads per k-step.
  const __bf16* arow  = X + (size_t)(mt * 16 + r) * K + hi * 8;
  const __bf16* bbase = Wp + (size_t)nt0 * Ksteps * 512 + lane * 16;
  __builtin_prefetch(arow, 0, 0);
  __builtin_prefetch(bbase, 0, 0);

  v8f acc0 = v8f{0.f,0.f,0.f,0.f,0.f,0.f,0.f,0.f};
  v8f acc1 = acc0, acc2 = acc0, acc3 = acc0;

  for (int ks = 0; ks < Ksteps; ++ks) {
    const u32x4* ap = (const u32x4*)(arow + ks * 32);
    v16bf a = ld_frag(ap + 0, ap + 2);
    const u32x4* bp0 = (const u32x4*)(bbase + (size_t)ks * 512);
    const u32x4* bp1 = (const u32x4*)(bbase + ((size_t)Ksteps     + ks) * 512);
    const u32x4* bp2 = (const u32x4*)(bbase + ((size_t)Ksteps * 2 + ks) * 512);
    const u32x4* bp3 = (const u32x4*)(bbase + ((size_t)Ksteps * 3 + ks) * 512);
    v16bf b0 = ld_frag(bp0 + 0, bp0 + 1);
    v16bf b1 = ld_frag(bp1 + 0, bp1 + 1);
    v16bf b2 = ld_frag(bp2 + 0, bp2 + 1);
    v16bf b3 = ld_frag(bp3 + 0, bp3 + 1);
    acc0 = __builtin_amdgcn_wmma_f32_16x16x32_bf16(false, a, false, b0, (short)0, acc0, false, false);
    acc1 = __builtin_amdgcn_wmma_f32_16x16x32_bf16(false, a, false, b1, (short)0, acc1, false, false);
    acc2 = __builtin_amdgcn_wmma_f32_16x16x32_bf16(false, a, false, b2, (short)0, acc2, false, false);
    acc3 = __builtin_amdgcn_wmma_f32_16x16x32_bf16(false, a, false, b3, (short)0, acc3, false, false);
  }

#pragma unroll
  for (int u = 0; u < 4; ++u) {
    const v8f acc = (u == 0) ? acc0 : (u == 1) ? acc1 : (u == 2) ? acc2 : acc3;
    const int n = (nt0 + u) * 16 + r;
    float wc0 = 0.f, wc1 = 0.f, wc2 = 0.f;
    if (Xc) { wc0 = Wc[n]; wc1 = Wc[Nout + n]; wc2 = Wc[2 * Nout + n]; }
    float bb = 0.f, gg = 0.f, ee = 0.f;
    if (n < Nout) { bb = bias[n]; gg = gain[n]; ee = beta[n]; }
#pragma unroll
    for (int vi = 0; vi < 8; ++vi) {
      const int m = mt * 16 + vi + hi * 8;  // D layout: lanes>=16 hold rows M+8
      float v = acc[vi];
      if (Xc) {
        const float* c = Xc + (size_t)m * 3;
        v = fmaf(c[0], wc0, v);
        v = fmaf(c[1], wc1, v);
        v = fmaf(c[2], wc2, v);
      }
      v = (v + bb) * gg + ee;
      v = v > 0.f ? v : 0.f;
      if (n < Nout) {
        if (Yf) Yf[(size_t)m * Ystride + n] = v;
        else    Yb[(size_t)m * Ystride + n] = (__bf16)v;
      }
    }
  }
}

// ---------------------------------------------------------------------------
// Max-pool over R consecutive rows per group: out[g,c] = max_r A[g*R+r, c].
// ---------------------------------------------------------------------------
__global__ __launch_bounds__(256)
void maxpool_kernel(const __bf16* __restrict__ A, int Gtot, int R, int C,
                    __bf16* __restrict__ out) {
  const int gid = blockIdx.x * 256 + threadIdx.x;
  if (gid >= Gtot * C) return;
  const int g = gid / C, c = gid % C;
  const __bf16* p = A + (size_t)g * R * C + c;
  float m = -3.0e38f;
  for (int r = 0; r < R; ++r) {
    float v = (float)p[(size_t)r * C];
    m = v > m ? v : m;
  }
  out[(size_t)g * C + c] = (__bf16)m;
}

// ---------------------------------------------------------------------------
// Host side
// ---------------------------------------------------------------------------
static inline void launch_gemm(hipStream_t st, const void* X, const void* Wp,
                               const float* b, const float* g, const float* be,
                               const float* Xc, const float* Wc,
                               int M, int K, int Nout, int Npad, int Ystride,
                               void* Yb, float* Yf) {
  long long waves = (long long)(M >> 4) * (Npad >> 6);
  int blocks = (int)((waves * 32 + 255) / 256);
  wmma_gemm_kernel<<<blocks, 256, 0, st>>>((const __bf16*)X, (const __bf16*)Wp,
      b, g, be, Xc, Wc, M, K, Nout, Npad, Ystride, (__bf16*)Yb, Yf);
}

static inline void launch_pack(hipStream_t st, const float* W, int K, int N,
                               int Npad, void* Wp) {
  int total = (Npad >> 4) * (K >> 5) * 512;
  int blocks = (total + 255) / 256;
  if (blocks > 4096) blocks = 4096;
  pack_w_kernel<<<blocks, 256, 0, st>>>(W, K, N, Npad, (__bf16*)Wp);
}

extern "C" void kernel_launch(void* const* d_in, const int* in_sizes, int n_in,
                              void* d_out, int out_size, void* d_ws, size_t ws_size,
                              hipStream_t stream) {
  (void)out_size; (void)ws_size;
  constexpr int B  = 32;
  constexpr int M0 = B * 512 * 32;   // 524288 rows, level-0 grouped points
  constexpr int M1 = B * 128 * 64;   // 262144 rows, level-1 grouped points
  constexpr int M2 = B * 128;        // 4096 rows, level-2 points

  // ---- input mapping (handle insertion-order or pytree-sorted flattening) --
  const bool xyz_first = (in_sizes[0] == B * 4096 * 3);
  const int xyzIdx = xyz_first ? 0 : (n_in - 1);
  int base_mlp0, base_mlp1, base_mlp2, base_fc;
  if (xyz_first) { base_mlp0 = 1;  base_mlp1 = 13; base_mlp2 = 25; base_fc = 37; }
  else           { base_fc = 0;    base_mlp0 = 12; base_mlp1 = 24; base_mlp2 = 36; }
  auto P = [&](int base, int layer, int comp) -> const float* {
    return (const float*)d_in[base + layer * 4 + comp];
  };
  const float* xyz = (const float*)d_in[xyzIdx];

  // ---- workspace carve ----------------------------------------------------
  char* ws = (char*)d_ws;
  size_t off = 0;
  auto alloc = [&](size_t bytes) -> char* {
    char* p = ws + off;
    off += (bytes + 255) & ~(size_t)255;
    return p;
  };
  float* p0  = (float*)alloc((size_t)B * 1024 * 3 * 4);
  float* p1  = (float*)alloc((size_t)B * 512 * 3 * 4);
  float* p2  = (float*)alloc((size_t)B * 128 * 3 * 4);
  int*   nb0 = (int*)  alloc((size_t)M0 * 4);
  int*   nb1 = (int*)  alloc((size_t)M1 * 4);
  float* Xc0 = (float*)alloc((size_t)M0 * 3 * 4);
  float* Xc1 = (float*)alloc((size_t)M1 * 3 * 4);
  __bf16* f0 = (__bf16*)alloc((size_t)B * 512 * 128 * 2);
  __bf16* X2 = (__bf16*)alloc((size_t)M2 * 256 * 2);
  __bf16* f2 = (__bf16*)alloc((size_t)B * 1024 * 2);
  __bf16* fcA = (__bf16*)alloc((size_t)B * 512 * 2);
  __bf16* fcB = (__bf16*)alloc((size_t)B * 256 * 2);

  auto packBytes = [](int K, int Npad) { return (size_t)(Npad / 16) * (K / 32) * 512 * 2; };
  void* Wp_l0w2 = alloc(packBytes(64, 64));
  void* Wp_l0w3 = alloc(packBytes(64, 128));
  void* Wp_l1w1 = alloc(packBytes(128, 128));
  void* Wp_l1w2 = alloc(packBytes(128, 128));
  void* Wp_l1w3 = alloc(packBytes(128, 256));
  void* Wp_l2w1 = alloc(packBytes(256, 256));
  void* Wp_l2w2 = alloc(packBytes(256, 512));
  void* Wp_l2w3 = alloc(packBytes(512, 1024));
  void* Wp_fc1  = alloc(packBytes(1024, 512));
  void* Wp_fc2  = alloc(packBytes(512, 256));
  void* Wp_fc3  = alloc(packBytes(256, 64));   // N padded 40 -> 64 (full strip)

  // Two 134MB ping-pong regions for big activations (carefully aliased).
  char* Ra = alloc((size_t)M0 * 128 * 2);
  char* Rb = alloc((size_t)M0 * 128 * 2);
  __bf16* A0a = (__bf16*)Ra;                                // [M0,64]
  __bf16* A0b = (__bf16*)(Ra + (size_t)M0 * 64 * 2);        // [M0,64]
  __bf16* A0c = (__bf16*)Rb;                                // [M0,128]
  __bf16* X1  = (__bf16*)Ra;                                // [M1,128] (A0a dead)
  __bf16* A1a = (__bf16*)(Ra + (size_t)M0 * 64 * 2);        // [M1,128] (A0b dead)
  __bf16* A1b = (__bf16*)Rb;                                // [M1,128] (A0c dead)
  __bf16* A1c = (__bf16*)Ra;                                // [M1,256] (X1,A1a dead)
  __bf16* A2a = (__bf16*)Rb;                                // [M2,256]
  __bf16* A2b = (__bf16*)(Rb + (size_t)M2 * 256 * 2);       // [M2,512]
  __bf16* A2c = (__bf16*)(Rb + (size_t)M2 * 256 * 2 + (size_t)M2 * 512 * 2); // [M2,1024]

  // ---- pack all weights (independent of geometry pipeline) ---------------
  launch_pack(stream, P(base_mlp0,1,0), 64, 64, 64, Wp_l0w2);
  launch_pack(stream, P(base_mlp0,2,0), 64, 128, 128, Wp_l0w3);
  launch_pack(stream, P(base_mlp1,0,0) + 3*128, 128, 128, 128, Wp_l1w1);  // rows 3..130
  launch_pack(stream, P(base_mlp1,1,0), 128, 128, 128, Wp_l1w2);
  launch_pack(stream, P(base_mlp1,2,0), 128, 256, 256, Wp_l1w3);
  launch_pack(stream, P(base_mlp2,0,0) + 3*256, 256, 256, 256, Wp_l2w1);  // rows 3..258
  launch_pack(stream, P(base_mlp2,1,0), 256, 512, 512, Wp_l2w2);
  launch_pack(stream, P(base_mlp2,2,0), 512, 1024, 1024, Wp_l2w3);
  launch_pack(stream, P(base_fc,0,0), 1024, 512, 512, Wp_fc1);
  launch_pack(stream, P(base_fc,1,0), 512, 256, 256, Wp_fc2);
  launch_pack(stream, P(base_fc,2,0), 256, 40, 64, Wp_fc3);               // N padded 40->64

  // ---- geometry: FPS chains + ball queries --------------------------------
  fps_kernel<<<B, 256, 0, stream>>>(xyz, 4096, 1024, p0);
  fps_kernel<<<B, 256, 0, stream>>>(p0, 1024, 512, p1);
  fps_kernel<<<B, 256, 0, stream>>>(p1, 512, 128, p2);
  ballq_kernel<<<dim3(512 / 8, B), 256, 0, stream>>>(p0, 1024, p1, 512, 32, 0.04f, nb0);
  ballq_kernel<<<dim3(128 / 8, B), 256, 0, stream>>>(p1, 512, p2, 128, 64, 0.16f, nb1);

  // ---- level 0: group -> (3->64 scalar) -> WMMA 64->64 -> 64->128 -> pool -
  group0_kernel<<<(M0 + 255) / 256, 256, 0, stream>>>(p0, p1, nb0, M0, Xc0);
  coord_mlp_kernel<<<(M0 * 64 + 255) / 256, 256, 0, stream>>>(
      Xc0, P(base_mlp0,0,0), P(base_mlp0,0,1), P(base_mlp0,0,2), P(base_mlp0,0,3), M0, A0a);
  launch_gemm(stream, A0a, Wp_l0w2, P(base_mlp0,1,1), P(base_mlp0,1,2), P(base_mlp0,1,3),
              nullptr, nullptr, M0, 64, 64, 64, 64, A0b, nullptr);
  launch_gemm(stream, A0b, Wp_l0w3, P(base_mlp0,2,1), P(base_mlp0,2,2), P(base_mlp0,2,3),
              nullptr, nullptr, M0, 64, 128, 128, 128, A0c, nullptr);
  maxpool_kernel<<<(B * 512 * 128 + 255) / 256, 256, 0, stream>>>(A0c, B * 512, 32, 128, f0);

  // ---- level 1: group -> WMMA (K=128 + coords) -> 128->128 -> 128->256 ----
  group1_kernel<<<(M1 * 32 + 255) / 256, 256, 0, stream>>>(p1, p2, f0, nb1, M1, X1, Xc1);
  launch_gemm(stream, X1, Wp_l1w1, P(base_mlp1,0,1), P(base_mlp1,0,2), P(base_mlp1,0,3),
              Xc1, P(base_mlp1,0,0), M1, 128, 128, 128, 128, A1a, nullptr);
  launch_gemm(stream, A1a, Wp_l1w2, P(base_mlp1,1,1), P(base_mlp1,1,2), P(base_mlp1,1,3),
              nullptr, nullptr, M1, 128, 128, 128, 128, A1b, nullptr);
  launch_gemm(stream, A1b, Wp_l1w3, P(base_mlp1,2,1), P(base_mlp1,2,2), P(base_mlp1,2,3),
              nullptr, nullptr, M1, 128, 256, 256, 256, A1c, nullptr);
  maxpool_kernel<<<(M2 * 256 + 255) / 256, 256, 0, stream>>>(A1c, M2, 64, 256, X2);

  // ---- level 2: WMMA (K=256 + raw p2 coords) -> 256->512 -> 512->1024 -----
  launch_gemm(stream, X2, Wp_l2w1, P(base_mlp2,0,1), P(base_mlp2,0,2), P(base_mlp2,0,3),
              p2, P(base_mlp2,0,0), M2, 256, 256, 256, 256, A2a, nullptr);
  launch_gemm(stream, A2a, Wp_l2w2, P(base_mlp2,1,1), P(base_mlp2,1,2), P(base_mlp2,1,3),
              nullptr, nullptr, M2, 256, 512, 512, 512, A2b, nullptr);
  launch_gemm(stream, A2b, Wp_l2w3, P(base_mlp2,2,1), P(base_mlp2,2,2), P(base_mlp2,2,3),
              nullptr, nullptr, M2, 512, 1024, 1024, 1024, A2c, nullptr);
  maxpool_kernel<<<(B * 1024 + 255) / 256, 256, 0, stream>>>(A2c, B, 128, 1024, f2);

  // ---- FC head: 1024->512->256->40 (final layer writes f32 to d_out) ------
  launch_gemm(stream, f2, Wp_fc1, P(base_fc,0,1), P(base_fc,0,2), P(base_fc,0,3),
              nullptr, nullptr, B, 1024, 512, 512, 512, fcA, nullptr);
  launch_gemm(stream, fcA, Wp_fc2, P(base_fc,1,1), P(base_fc,1,2), P(base_fc,1,3),
              nullptr, nullptr, B, 512, 256, 256, 256, fcB, nullptr);
  launch_gemm(stream, fcB, Wp_fc3, P(base_fc,2,1), P(base_fc,2,2), P(base_fc,2,3),
              nullptr, nullptr, B, 256, 40, 64, 40, nullptr, (float*)d_out);
}
